// AttnRNNModel_58428735094963
// MI455X (gfx1250) — matrix operand order repaired and checked by reading, not compiled, not run on hardware
//
#include <hip/hip_runtime.h>
#include <cstdint>
#include <cstddef>

// ---------------- CDNA5 WMMA types & helpers (wave32) ----------------
typedef __attribute__((ext_vector_type(16))) __bf16 v16bf;
typedef __attribute__((ext_vector_type(8)))  __bf16 v8bf;
typedef __attribute__((ext_vector_type(8)))  float  v8f;

__device__ __forceinline__ float bf2f(__bf16 x) {
  unsigned short u = __builtin_bit_cast(unsigned short, x);
  unsigned int v = ((unsigned int)u) << 16;
  return __builtin_bit_cast(float, v);
}
__device__ __forceinline__ __bf16 f2bf(float x) {
  unsigned int v = __builtin_bit_cast(unsigned int, x);
  unsigned int r = v + 0x7FFFu + ((v >> 16) & 1u);   // round-to-nearest-even
  unsigned short u = (unsigned short)(r >> 16);
  return __builtin_bit_cast(__bf16, u);
}

__device__ __forceinline__ v8f wmma_bf16(v16bf a, v16bf b, v8f c) {
  // D = A(16x32) * B(32x16) + C, f32 accumulate
  return __builtin_amdgcn_wmma_f32_16x16x32_bf16(false, a, false, b, (short)0, c,
                                                 false, false);
}

// A fragment: 16x32 bf16 tile, row-major source with leading dim lda (elements).
// Lane layout (ISA 7.12.2): lanes 0-15 row M=lane, K chunks [0,8)+[16,24);
// lanes 16-31 row M=lane-16, K chunks [8,16)+[24,32).
__device__ __forceinline__ v16bf load_a_frag(const __bf16* __restrict__ base, int lda) {
  int lane = threadIdx.x & 31;
  int m  = lane & 15;
  int kh = (lane >> 4) << 3;            // 0 or 8
  const __bf16* p0 = base + (size_t)m * lda + kh;
  v8bf x0 = *(const v8bf*)p0;
  v8bf x1 = *(const v8bf*)(p0 + 16);
  v16bf a;
#pragma unroll
  for (int i = 0; i < 8; ++i) { a[i] = x0[i]; a[i + 8] = x1[i]; }
  return a;
}

// B fragment: 32x16 (KxN) where B[k][n] = W[n][k]; W is (N,K) row-major bf16.
// Lane n holds column n: lanes 0-15 K=[0,8)+[16,24), lanes 16-31 K=[8,16)+[24,32).
__device__ __forceinline__ v16bf load_b_frag(const __bf16* __restrict__ wBase, int ldb) {
  int lane = threadIdx.x & 31;
  int n  = lane & 15;
  int ks = (lane >> 4) << 3;            // 0 or 8
  const __bf16* p0 = wBase + (size_t)n * ldb + ks;
  v8bf x0 = *(const v8bf*)p0;
  v8bf x1 = *(const v8bf*)(p0 + 16);
  v16bf b;
#pragma unroll
  for (int i = 0; i < 8; ++i) { b[i] = x0[i]; b[i + 8] = x1[i]; }
  return b;
}

// ---------------- elementwise fp32 -> bf16 weight conversion ----------------
__global__ void k_f2bf(const float* __restrict__ x, __bf16* __restrict__ y, int n) {
  int i = blockIdx.x * blockDim.x + threadIdx.x;
  if (i < n) y[i] = f2bf(x[i]);
}

// ---------------- embedding gather (fp32 table -> bf16 rows) ----------------
__global__ void k_gather_embed(const float* __restrict__ emb, const int* __restrict__ tok,
                               __bf16* __restrict__ out, int D) {
  int r = blockIdx.x;
  const float* src = emb + (size_t)tok[r] * D;
  __bf16* dst = out + (size_t)r * D;
  for (int j = threadIdx.x; j < D; j += blockDim.x) dst[j] = f2bf(src[j]);
}

// ---------------- gather enc_hid = enc_out[b, src_len[b]-1, :] ----------------
__global__ void k_gather_hid(const __bf16* __restrict__ encY, const int* __restrict__ src_len,
                             __bf16* __restrict__ out, int S, int H) {
  int b = blockIdx.x;
  int sl = src_len[b] - 1;
  const __bf16* src = encY + ((size_t)b * S + sl) * H;
  __bf16* dst = out + (size_t)b * H;
  for (int j = threadIdx.x; j < H; j += blockDim.x) dst[j] = src[j];
}

// ---------------- generic WMMA GEMM: out = act(A(M,K)bf16 @ W(N,K)^T + b) -------
// Block: 256 thr = 8 waves; tile M=16, N=128 (16 per wave). A K-panel staged in LDS.
__global__ __launch_bounds__(256) void k_gemm(
    const __bf16* __restrict__ A, const __bf16* __restrict__ W,
    const float* __restrict__ bias, float* __restrict__ outF,
    __bf16* __restrict__ outB, int M, int N, int K, int act) {
  __shared__ __bf16 sA[16 * 1024];      // up to K=1024
  int m0 = blockIdx.y << 4;
  int n0 = (blockIdx.x << 7) + ((threadIdx.x >> 5) << 4);

  // stage contiguous 16xK A panel
  {
    const uint32_t* src = (const uint32_t*)(A + (size_t)m0 * K);
    uint32_t* dst = (uint32_t*)sA;
    int words = (16 * K) >> 1;
    for (int i = threadIdx.x; i < words; i += 256) dst[i] = src[i];
  }
  __syncthreads();

  const __bf16* wRow = W + (size_t)n0 * K;
  v8f c = {};
  for (int kk = 0; kk < K; kk += 32) {
    __builtin_prefetch(wRow + kk + 64, 0, 0);          // global_prefetch_b8
    v16bf a = load_a_frag(sA + kk, K);
    v16bf b = load_b_frag(wRow + kk, K);
    c = wmma_bf16(a, b, c);
  }

  int lane = threadIdx.x & 31;
  int n  = lane & 15;
  int rb = (lane >> 4) << 3;
  float bv = bias ? bias[n0 + n] : 0.0f;
#pragma unroll
  for (int r = 0; r < 8; ++r) {
    float v = c[r] + bv;
    if (act == 1) v = fmaxf(v, 0.0f);
    size_t off = (size_t)(m0 + rb + r) * N + (n0 + n);
    if (outF) outF[off] = v;
    if (outB) outB[off] = f2bf(v);
  }
}

// ---------------- encoder tanh-RNN scan (persistent single workgroup) ----------
// h_t = tanh(x_t @ Wih^T + bih + h @ Whh^T + bhh); x rows are (b*S + t).
// Block: 1024 thr = 32 waves, one 16x16 N-tile each (N = 512). h lives in LDS.
__global__ __launch_bounds__(1024) void k_rnn_scan(
    const __bf16* __restrict__ X, const __bf16* __restrict__ Wih,
    const __bf16* __restrict__ Whh, const float* __restrict__ bih,
    const float* __restrict__ bhh, __bf16* __restrict__ Y,
    float* __restrict__ Yf, int S, int H) {
  __shared__ __bf16 sH[16 * 512];
  for (int i = threadIdx.x; i < 16 * 512; i += blockDim.x) sH[i] = f2bf(0.0f);
  __syncthreads();

  int n0   = (threadIdx.x >> 5) << 4;
  int lane = threadIdx.x & 31;
  int n    = lane & 15;
  int rb   = (lane >> 4) << 3;
  float bsum = bih[n0 + n] + bhh[n0 + n];
  const __bf16* wi = Wih + (size_t)n0 * H;
  const __bf16* wh = Whh + (size_t)n0 * H;

  for (int t = 0; t < S; ++t) {
    const __bf16* xt = X + (size_t)t * H;       // row m at stride S*H
    v8f c = {};
    for (int kk = 0; kk < H; kk += 32) {
      v16bf ax = load_a_frag(xt + kk, S * H);
      v16bf bi = load_b_frag(wi + kk, H);
      c = wmma_bf16(ax, bi, c);
      v16bf ah = load_a_frag(sH + kk, 512);
      v16bf bh = load_b_frag(wh + kk, H);
      c = wmma_bf16(ah, bh, c);
    }
    __syncthreads();                             // all reads of sH done
#pragma unroll
    for (int r = 0; r < 8; ++r) {
      int m = rb + r;
      float v = tanhf(c[r] + bsum);
      __bf16 bv = f2bf(v);
      sH[m * 512 + n0 + n] = bv;
      size_t off = ((size_t)m * S + t) * H + (n0 + n);
      Y[off] = bv;
      if (Yf) Yf[off] = v;
    }
    __syncthreads();
  }
}

// ---------------- decoder attention-RNN scan (persistent single workgroup) -----
__global__ __launch_bounds__(1024) void k_attn_rnn_scan(
    const __bf16* __restrict__ X, const __bf16* __restrict__ H0,
    const float* __restrict__ encOut, const __bf16* __restrict__ Wih,
    const __bf16* __restrict__ Whh, const float* __restrict__ bih,
    const float* __restrict__ bhh, const __bf16* __restrict__ Wa,
    const float* __restrict__ ba, __bf16* __restrict__ Y, int T, int S, int H) {
  __shared__ __bf16 sH[16 * 512];
  __shared__ float  sScore[16 * 128];
  __shared__ __bf16 sAttn[16 * 512];
  __shared__ __bf16 sXin[16 * 512];
  for (int i = threadIdx.x; i < 16 * 512; i += blockDim.x) sH[i] = H0[i];
  __syncthreads();

  int n0   = (threadIdx.x >> 5) << 4;
  int lane = threadIdx.x & 31;
  int n    = lane & 15;
  int rb   = (lane >> 4) << 3;
  float bsum = bih[n0 + n] + bhh[n0 + n];
  float bav  = ba[n0 + n];
  const __bf16* wi = Wih + (size_t)n0 * H;
  const __bf16* wh = Whh + (size_t)n0 * H;
  const __bf16* wa = Wa  + (size_t)n0 * H;

  for (int t = 0; t < T; ++t) {
    // 1) scores[b,s] = <h_b, enc[b,s,:]>
    for (int p = threadIdx.x; p < 16 * 128; p += blockDim.x) {
      int b = p >> 7, s = p & 127;
      const float*  e = encOut + ((size_t)b * S + s) * H;
      const __bf16* h = sH + b * 512;
      float acc = 0.0f;
      for (int j = 0; j < H; ++j) acc += bf2f(h[j]) * e[j];
      sScore[p] = acc;
    }
    __syncthreads();
    // 2) softmax over s (one thread per batch row)
    if (threadIdx.x < 16) {
      float* row = sScore + threadIdx.x * 128;
      float mx = row[0];
      for (int s = 1; s < 128; ++s) mx = fmaxf(mx, row[s]);
      float sum = 0.0f;
      for (int s = 0; s < 128; ++s) { float e = __expf(row[s] - mx); row[s] = e; sum += e; }
      float inv = 1.0f / sum;
      for (int s = 0; s < 128; ++s) row[s] *= inv;
    }
    __syncthreads();
    // 3) attn[b,j] = sum_s score * enc[b,s,j]
    for (int p = threadIdx.x; p < 16 * 512; p += blockDim.x) {
      int b = p >> 9, j = p & 511;
      const float* rowS = sScore + b * 128;
      const float* e = encOut + (size_t)b * S * H + j;
      float acc = 0.0f;
      for (int s = 0; s < S; ++s) acc += rowS[s] * e[(size_t)s * H];
      sAttn[p] = f2bf(acc);
    }
    __syncthreads();
    // 4) xin = x_t + attn @ Wa^T + ba   (WMMA)
    {
      v8f c = {};
      for (int kk = 0; kk < H; kk += 32) {
        v16bf a = load_a_frag(sAttn + kk, 512);
        v16bf b = load_b_frag(wa + kk, H);
        c = wmma_bf16(a, b, c);
      }
      const __bf16* xt = X + (size_t)t * H;     // row m at stride T*H
#pragma unroll
      for (int r = 0; r < 8; ++r) {
        int m = rb + r;
        float xv = bf2f(xt[(size_t)m * T * H + (n0 + n)]);
        sXin[m * 512 + n0 + n] = f2bf(xv + bav + c[r]);
      }
    }
    __syncthreads();
    // 5) h' = tanh(xin @ Wih^T + h @ Whh^T + b)   (WMMA)
    v8f c = {};
    for (int kk = 0; kk < H; kk += 32) {
      v16bf a1 = load_a_frag(sXin + kk, 512);
      v16bf b1 = load_b_frag(wi + kk, H);
      c = wmma_bf16(a1, b1, c);
      v16bf a2 = load_a_frag(sH + kk, 512);
      v16bf b2 = load_b_frag(wh + kk, H);
      c = wmma_bf16(a2, b2, c);
    }
    __syncthreads();
#pragma unroll
    for (int r = 0; r < 8; ++r) {
      int m = rb + r;
      float v = tanhf(c[r] + bsum);
      __bf16 bv = f2bf(v);
      sH[m * 512 + n0 + n] = bv;
      Y[((size_t)m * T + t) * H + (n0 + n)] = bv;
    }
    __syncthreads();
  }
}

// ================================ host side ================================
namespace {
constexpr int B = 16, S = 128, T = 64, V = 32000, D = 512, H = 512, L = 2;

inline size_t alignUp(size_t x) { return (x + 255) & ~(size_t)255; }

inline void cvt(const void* src, __bf16* dst, int n, hipStream_t st) {
  k_f2bf<<<(n + 255) / 256, 256, 0, st>>>((const float*)src, dst, n);
}
}  // namespace

extern "C" void kernel_launch(void* const* d_in, const int* in_sizes, int n_in,
                              void* d_out, int out_size, void* d_ws, size_t ws_size,
                              hipStream_t stream) {
  // ---- input pointers (setup_inputs dict order, params flattened) ----
  const int*   src      = (const int*)d_in[0];
  const int*   src_len  = (const int*)d_in[1];
  const int*   tgt      = (const int*)d_in[2];
  const float* enc_emb  = (const float*)d_in[3];
  const float* emlp_w1  = (const float*)d_in[4];
  const float* emlp_b1  = (const float*)d_in[5];
  const float* emlp_w2  = (const float*)d_in[6];
  const float* emlp_b2  = (const float*)d_in[7];
  const float* ernn_wih = (const float*)d_in[8];
  const float* ernn_whh = (const float*)d_in[9];
  const float* ernn_bih = (const float*)d_in[10];
  const float* ernn_bhh = (const float*)d_in[11];
  const float* dec_emb  = (const float*)d_in[12];
  const float* dmlp_w1  = (const float*)d_in[13];
  const float* dmlp_b1  = (const float*)d_in[14];
  const float* dmlp_w2  = (const float*)d_in[15];
  const float* dmlp_b2  = (const float*)d_in[16];
  const float* e2h_w1   = (const float*)d_in[17];
  const float* e2h_b1   = (const float*)d_in[18];
  const float* e2h_w2   = (const float*)d_in[19];
  const float* e2h_b2   = (const float*)d_in[20];
  const float* attn_w   = (const float*)d_in[21];
  const float* attn_b   = (const float*)d_in[22];
  const float* drnn_wih = (const float*)d_in[23];
  const float* drnn_whh = (const float*)d_in[24];
  const float* drnn_bih = (const float*)d_in[25];
  const float* drnn_bhh = (const float*)d_in[26];
  const float* h2e_w1   = (const float*)d_in[27];
  const float* h2e_b1   = (const float*)d_in[28];
  const float* h2e_w2   = (const float*)d_in[29];
  const float* h2e_b2   = (const float*)d_in[30];

  // ---- workspace suballocation (256B aligned) ----
  uint8_t* ws = (uint8_t*)d_ws;
  size_t cur = 0;
  auto allocBf = [&](size_t elems) -> __bf16* {
    __bf16* p = (__bf16*)(ws + cur);
    cur = alignUp(cur + elems * sizeof(__bf16));
    return p;
  };
  auto allocF = [&](size_t elems) -> float* {
    float* p = (float*)(ws + cur);
    cur = alignUp(cur + elems * sizeof(float));
    return p;
  };

  const size_t HH = (size_t)H * H;                 // 262144
  __bf16* wEmlp1  = allocBf(HH);
  __bf16* wEmlp2  = allocBf(HH);
  __bf16* wEWih   = allocBf(L * HH);
  __bf16* wEWhh   = allocBf(L * HH);
  __bf16* wDmlp1  = allocBf(HH);
  __bf16* wDmlp2  = allocBf(HH);
  __bf16* wE2h1   = allocBf((size_t)(L * H) * H);  // 1024x512
  __bf16* wE2h2   = allocBf((size_t)(L * H) * (L * H));  // 1024x1024
  __bf16* wAttn   = allocBf(HH);
  __bf16* wDWih   = allocBf(L * HH);
  __bf16* wDWhh   = allocBf(L * HH);
  __bf16* wH2e1   = allocBf(HH);
  __bf16* wH2e2   = allocBf((size_t)D * H);
  __bf16* wDecEmb = allocBf((size_t)V * D);        // 32000x512

  __bf16* xEnc   = allocBf((size_t)B * S * D);     // 2048x512
  __bf16* tmpAct = allocBf((size_t)B * S * H);     // reused enc/dec MLP mid
  __bf16* encA   = allocBf((size_t)B * S * H);
  __bf16* encB   = allocBf((size_t)B * S * H);
  float*  encOutF= allocF((size_t)B * S * H);      // fp32 enc_out for attention
  __bf16* encHid = allocBf((size_t)B * H);
  __bf16* e2hT   = allocBf((size_t)B * L * H);     // 16x1024
  __bf16* e2hOut = allocBf((size_t)B * L * H);     // 16x1024 (flat; h0_l at +l*8192)
  __bf16* xDec   = allocBf((size_t)B * T * D);     // 1024x512
  __bf16* decA   = allocBf((size_t)B * T * H);
  __bf16* decB   = allocBf((size_t)B * T * H);
  __bf16* h2eT   = allocBf((size_t)B * T * H);
  __bf16* yBuf   = allocBf((size_t)B * T * D);
  (void)ws_size; (void)n_in; (void)in_sizes; (void)out_size;

  // ---- 0) convert all weights fp32 -> bf16 ----
  cvt(emlp_w1, wEmlp1, HH, stream);
  cvt(emlp_w2, wEmlp2, HH, stream);
  cvt(ernn_wih, wEWih, L * HH, stream);
  cvt(ernn_whh, wEWhh, L * HH, stream);
  cvt(dmlp_w1, wDmlp1, HH, stream);
  cvt(dmlp_w2, wDmlp2, HH, stream);
  cvt(e2h_w1, wE2h1, (L * H) * H, stream);
  cvt(e2h_w2, wE2h2, (L * H) * (L * H), stream);
  cvt(attn_w, wAttn, HH, stream);
  cvt(drnn_wih, wDWih, L * HH, stream);
  cvt(drnn_whh, wDWhh, L * HH, stream);
  cvt(h2e_w1, wH2e1, HH, stream);
  cvt(h2e_w2, wH2e2, D * H, stream);
  cvt(dec_emb, wDecEmb, (size_t)V * D, stream);

  // ---- 1) encoder: embed -> MLP(relu,relu) ----
  k_gather_embed<<<B * S, 256, 0, stream>>>(enc_emb, src, xEnc, D);
  k_gemm<<<dim3(H / 128, (B * S) / 16), 256, 0, stream>>>(
      xEnc, wEmlp1, emlp_b1, nullptr, tmpAct, B * S, H, D, 1);
  k_gemm<<<dim3(H / 128, (B * S) / 16), 256, 0, stream>>>(
      tmpAct, wEmlp2, emlp_b2, nullptr, encA, B * S, H, H, 1);

  // ---- 2) encoder RNN layers (persistent scan, h in LDS) ----
  k_rnn_scan<<<1, 1024, 0, stream>>>(encA, wEWih, wEWhh, ernn_bih, ernn_bhh,
                                     encB, nullptr, S, H);
  k_rnn_scan<<<1, 1024, 0, stream>>>(encB, wEWih + HH, wEWhh + HH,
                                     ernn_bih + H, ernn_bhh + H,
                                     encA, encOutF, S, H);

  // ---- 3) enc_hid gather + e2h MLP (relu,relu); reshape(L,-1,H) == flat+l*B*H ----
  k_gather_hid<<<B, 256, 0, stream>>>(encA, src_len, encHid, S, H);
  k_gemm<<<dim3((L * H) / 128, B / 16), 256, 0, stream>>>(
      encHid, wE2h1, e2h_b1, nullptr, e2hT, B, L * H, H, 1);
  k_gemm<<<dim3((L * H) / 128, B / 16), 256, 0, stream>>>(
      e2hT, wE2h2, e2h_b2, nullptr, e2hOut, B, L * H, L * H, 1);

  // ---- 4) decoder: embed -> MLP(relu,relu) ----
  k_gather_embed<<<B * T, 256, 0, stream>>>(dec_emb, tgt, xDec, D);
  k_gemm<<<dim3(H / 128, (B * T) / 16), 256, 0, stream>>>(
      xDec, wDmlp1, dmlp_b1, nullptr, tmpAct, B * T, H, D, 1);
  k_gemm<<<dim3(H / 128, (B * T) / 16), 256, 0, stream>>>(
      tmpAct, wDmlp2, dmlp_b2, nullptr, decA, B * T, H, H, 1);

  // ---- 5) decoder attention-RNN layers ----
  k_attn_rnn_scan<<<1, 1024, 0, stream>>>(
      decA, e2hOut, encOutF, wDWih, wDWhh, drnn_bih, drnn_bhh,
      wAttn, attn_b, decB, T, S, H);
  k_attn_rnn_scan<<<1, 1024, 0, stream>>>(
      decB, e2hOut + (size_t)B * H, encOutF, wDWih + HH, wDWhh + HH,
      drnn_bih + H, drnn_bhh + H, wAttn, attn_b, decA, T, S, H);

  // ---- 6) h2e head: relu(lin) -> lin ----
  k_gemm<<<dim3(H / 128, (B * T) / 16), 256, 0, stream>>>(
      decA, wH2e1, h2e_b1, nullptr, h2eT, B * T, H, H, 1);
  k_gemm<<<dim3(D / 128, (B * T) / 16), 256, 0, stream>>>(
      h2eT, wH2e2, h2e_b2, nullptr, yBuf, B * T, D, H, 0);

  // ---- 7) tied-embedding logits: (B*T,D) @ dec_emb^T -> d_out fp32 ----
  k_gemm<<<dim3(V / 128, (B * T) / 16), 256, 0, stream>>>(
      yBuf, wDecEmb, nullptr, (float*)d_out, nullptr, B * T, V, D, 0);
}
